// model_1279900254287
// MI455X (gfx1250) — matrix operand order repaired and checked by reading, not compiled, not run on hardware
//
#include <hip/hip_runtime.h>
#include <hip/hip_bf16.h>
#include <math.h>

// FNO on gfx1250. Spectral conv recast as truncated-DFT GEMMs on
// v_wmma_f32_16x16x32_bf16. Memory-bound -> bf16 activations (halves HBM
// traffic vs fp32), f32 accumulation everywhere, fused epilogues.
// Fragment loads are packed: A = 2x b128, B = 8x b32 via pair-interleaved
// operand layouts that we control end-to-end.

typedef __attribute__((ext_vector_type(16))) __bf16        v16bf;
typedef __attribute__((ext_vector_type(8)))  float         v8f;
typedef __attribute__((ext_vector_type(8)))  unsigned int  v8u;
typedef __attribute__((ext_vector_type(4)))  unsigned int  v4u;

#define B_    8
#define C_    64
#define PIX_  65536          // 256*256

__device__ inline v8f wmma_bf16(v16bf a, v16bf b, v8f c) {
  return __builtin_amdgcn_wmma_f32_16x16x32_bf16(false, a, false, b, (short)0, c, false, false);
}

// ---- packed fragment loads (own k order: k = 16*(lane/16) + s for A and B) ---------------------
// A (16x32 MxK), row-major, 16B-aligned rows: two b128 loads per lane.
__device__ inline v16bf frag_A(const __bf16* t, int ld) {
  int l = threadIdx.x & 31;
  const __bf16* row = t + (l & 15) * ld + ((l & 16) ? 16 : 0);
  v4u a = *(const v4u*)(row);        // k = kb+0..7
  v4u b = *(const v4u*)(row + 8);    // k = kb+8..15
  v8u u;
  u[0] = a[0]; u[1] = a[1]; u[2] = a[2]; u[3] = a[3];
  u[4] = b[0]; u[5] = b[1]; u[6] = b[2]; u[7] = b[3];
  return __builtin_bit_cast(v16bf, u);
}
// B (32x16 KxN), PAIR-INTERLEAVED source: addr(k,n) = (k>>1)*twoN + 2n + (k&1)
__device__ inline v16bf frag_B_pi(const __bf16* t, int twoN) {
  int l = threadIdx.x & 31;
  const __bf16* p = t + ((l & 16) ? 8 * twoN : 0) + 2 * (l & 15);
  v8u u;
#pragma unroll
  for (int v = 0; v < 8; ++v) u[v] = *(const unsigned int*)(p + (size_t)v * twoN);
  return __builtin_bit_cast(v16bf, u);
}
// B (32x16 KxN), row-major source (u16 gather) -- used on async-staged LDS
__device__ inline v16bf frag_B_rm(const __bf16* t, int ld) {
  int l = threadIdx.x & 31;
  int n = l & 15;
  int kb = (l & 16) ? 16 : 0;
  v16bf f;
#pragma unroll
  for (int v = 0; v < 8; ++v) {
    f[2 * v]     = t[(kb + 2 * v) * ld + n];
    f[2 * v + 1] = t[(kb + 2 * v + 1) * ld + n];
  }
  return f;
}

// cooperative f32 -> bf16 row-major tile stage (single-wave workgroup)
template <int R, int C>
__device__ inline void stage_f32(__bf16* dst, const float* src, int srcStride) {
#pragma unroll
  for (int i = threadIdx.x; i < R * C; i += 32) {
    int r = i / C, c = i % C;
    dst[i] = (__bf16)src[r * srcStride + c];
  }
}

// ---- basis tables ----------------------------------------------------------
// Fwf: fwd-W B-panel [K=256(w)][N=32], pair-interleaved. cols<16: cos, >=16: -sin
// Ch/Sh/Shn: fwd-H A-panels [16(kx)][256(h)] row-major.
// Hi1/Hi2: inv-H A-panels [256(h)][32], (cos|-sin)/(HW) and (sin|cos)/(HW).
// Wi: inv-W B-panel [K=32][N=256(w)], pair-interleaved. k<16: c_k cos, else -c_k sin
__global__ void k_init(__bf16* Fwf, __bf16* Ch, __bf16* Sh, __bf16* Shn,
                       __bf16* Hi1, __bf16* Hi2, __bf16* Wi) {
  const float TP = 6.283185307179586f;
  for (int i = threadIdx.x; i < 8192; i += 256) {
    int w = i >> 5, k = i & 31, kk = k & 15;
    float th = TP * (float)(kk * w) / 256.f;
    float v = (k < 16) ? cosf(th) : -sinf(th);
    Fwf[(w >> 1) * 64 + 2 * k + (w & 1)] = (__bf16)v;
  }
  for (int i = threadIdx.x; i < 4096; i += 256) {
    int kx = i >> 8, h = i & 255;
    float th = TP * (float)(kx * h) / 256.f;
    Ch[i]  = (__bf16)cosf(th);
    Sh[i]  = (__bf16)sinf(th);
    Shn[i] = (__bf16)(-sinf(th));
  }
  const float s = 1.f / 65536.f;  // 1/(H*W)
  for (int i = threadIdx.x; i < 8192; i += 256) {
    int h = i >> 5, k = i & 31, kk = k & 15;
    float th = TP * (float)(kk * h) / 256.f;
    Hi1[i] = (__bf16)((((k < 16) ? cosf(th) : -sinf(th))) * s);
    Hi2[i] = (__bf16)((((k < 16) ? sinf(th) : cosf(th))) * s);
  }
  for (int i = threadIdx.x; i < 8192; i += 256) {
    int k = i >> 8, w = i & 255, kk = k & 15;
    float c = (kk == 0) ? 1.f : 2.f;
    float th = TP * (float)(kk * w) / 256.f;
    float v = (k < 16) ? c * cosf(th) : -c * sinf(th);
    Wi[(k >> 1) * 512 + 2 * w + (k & 1)] = (__bf16)v;
  }
}

// ---- fc0: [8,3,HW] f32 -> [8,64,HW] bf16 ----------------------------------
__global__ void k_fc0(const float* __restrict__ x, const float* __restrict__ w,
                      const float* __restrict__ bias, __bf16* __restrict__ h) {
  size_t idx = (size_t)blockIdx.x * 256 + threadIdx.x;  // 8*64*65536
  int p = (int)(idx & 65535);
  int c = (int)((idx >> 16) & 63);
  int b = (int)(idx >> 22);
  const float* xb = x + (size_t)b * 3 * PIX_ + p;
  float acc = bias[c] + w[c * 3 + 0] * xb[0] + w[c * 3 + 1] * xb[PIX_] +
              w[c * 3 + 2] * xb[2 * PIX_];
  h[idx] = (__bf16)acc;
}

// ---- forward DFT over W: rows (b,c,h), K=256 -> G panels (pair-interleaved)
__global__ void k_fwdW(const __bf16* __restrict__ hin, const __bf16* __restrict__ Fwf,
                       __bf16* __restrict__ G) {
  int row0 = blockIdx.x << 4;                       // 8192 blocks
  const __bf16* src = hin + (size_t)row0 * 256;
  v8f acc0 = {}, acc1 = {};
#pragma unroll
  for (int kc = 0; kc < 8; ++kc) {
    v16bf a  = frag_A(src + kc * 32, 256);          // direct bf16 vector loads
    v16bf b0 = frag_B_pi(Fwf + kc * 1024, 64);
    v16bf b1 = frag_B_pi(Fwf + kc * 1024 + 32, 64);
    acc0 = wmma_bf16(a, b0, acc0);
    acc1 = wmma_bf16(a, b1, acc1);
  }
  int l = threadIdx.x & 31, n = l & 15, mh = (l & 16) ? 8 : 0;
  int bc = row0 >> 8, hbase = row0 & 255;
  __bf16* gp = G + (size_t)bc * 8192;               // per-bc 256x32 panel, PI
#pragma unroll
  for (int r = 0; r < 8; ++r) {
    int h = hbase + r + mh;
    gp[(h >> 1) * 64 + 2 * n + (h & 1)]        = (__bf16)acc0[r];
    gp[(h >> 1) * 64 + 2 * (16 + n) + (h & 1)] = (__bf16)acc1[r];
  }
}

// ---- forward DFT over H: per (b,c), X[kx][ky] complex f32, K=256 ----------
__global__ void k_fwdH(const __bf16* __restrict__ G, const __bf16* __restrict__ Ch,
                       const __bf16* __restrict__ Sh, const __bf16* __restrict__ Shn,
                       float* __restrict__ X) {
  int bc = blockIdx.x;                              // 512 blocks
  const __bf16* g = G + (size_t)bc * 8192;
  v8f accR = {}, accI = {};
#pragma unroll
  for (int kc = 0; kc < 8; ++kc) {
    v16bf aC  = frag_A(Ch  + kc * 32, 256);
    v16bf aS  = frag_A(Sh  + kc * 32, 256);
    v16bf aSn = frag_A(Shn + kc * 32, 256);
    v16bf bR  = frag_B_pi(g + kc * 1024, 64);
    v16bf bI  = frag_B_pi(g + kc * 1024 + 32, 64);
    accR = wmma_bf16(aC, bR, accR);   // cos*Gr
    accR = wmma_bf16(aS, bI, accR);   // + sin*Gi
    accI = wmma_bf16(aC, bI, accI);   // cos*Gi
    accI = wmma_bf16(aSn, bR, accI);  // - sin*Gr
  }
  int l = threadIdx.x & 31, n = l & 15, mh = (l & 16) ? 8 : 0;
#pragma unroll
  for (int r = 0; r < 8; ++r) {
    int kx = r + mh;
    X[(size_t)bc * 512 + kx * 16 + n]       = accR[r];
    X[(size_t)bc * 512 + 256 + kx * 16 + n] = accI[r];
  }
}

// ---- per-mode 64x64 complex channel mix -> Xh panels (pair-interleaved) ---
__global__ void k_mix(const float* __restrict__ X, const float* __restrict__ wre,
                      const float* __restrict__ wim, __bf16* __restrict__ Xh) {
  int idx = blockIdx.x * 256 + threadIdx.x;         // 8*64*256
  int mode = idx & 255;
  int o = (idx >> 8) & 63;
  int b = idx >> 14;
  float orr = 0.f, oii = 0.f;
  for (int i = 0; i < 64; ++i) {
    float xr = X[(size_t)(b * 64 + i) * 512 + mode];
    float xi = X[(size_t)(b * 64 + i) * 512 + 256 + mode];
    float wr = wre[(size_t)(i * 64 + o) * 256 + mode];  // w indexed [i][o][mode]
    float wi = wim[(size_t)(i * 64 + o) * 256 + mode];
    orr += xr * wr - xi * wi;
    oii += xr * wi + xi * wr;
  }
  int kx = mode >> 4, ky = mode & 15;
  __bf16* xp = Xh + (size_t)(b * 64 + o) * 512;     // 32x16 panel, PI
  int kr = kx, ki = 16 + kx;
  xp[(kr >> 1) * 32 + 2 * ky + (kr & 1)] = (__bf16)orr;
  xp[(ki >> 1) * 32 + 2 * ky + (ki & 1)] = (__bf16)oii;
}

// ---- inverse DFT over H: Y[b,c,h][Yr|Yi] row-major bf16, K=32 -------------
__global__ void k_invH(const __bf16* __restrict__ Xh, const __bf16* __restrict__ Hi1,
                       const __bf16* __restrict__ Hi2, __bf16* __restrict__ Y) {
  int bc = blockIdx.x >> 4;                         // 8192 blocks
  int h0 = (blockIdx.x & 15) << 4;
  v16bf bX = frag_B_pi(Xh + (size_t)bc * 512, 32);
  v8f aR = {}, aI = {};
  aR = wmma_bf16(frag_A(Hi1 + h0 * 32, 32), bX, aR);
  aI = wmma_bf16(frag_A(Hi2 + h0 * 32, 32), bX, aI);
  int l = threadIdx.x & 31, n = l & 15, mh = (l & 16) ? 8 : 0;
#pragma unroll
  for (int r = 0; r < 8; ++r) {
    size_t row = (size_t)bc * 256 + h0 + r + mh;
    Y[row * 32 + n]      = (__bf16)aR[r];
    Y[row * 32 + 16 + n] = (__bf16)aI[r];
  }
}

// ---- conv1x1 skip: hout = W(64x64) * hin + bias (bf16 activations) --------
__global__ void k_conv(const __bf16* __restrict__ hin, const float* __restrict__ w,
                       const float* __restrict__ bias, __bf16* __restrict__ hout) {
  __shared__ __align__(16) __bf16 sA[16 * 32];
  __shared__ __align__(16) __bf16 sB[32 * 16];      // pair-interleaved
  int blk = blockIdx.x;                             // 131072 blocks
  int p0 = (blk & 4095) << 4;
  int o0 = ((blk >> 12) & 3) << 4;
  int b  = blk >> 14;
  const __bf16* hb = hin + (size_t)b * 64 * PIX_;
  __builtin_prefetch(hb + (size_t)(32 + (threadIdx.x & 31)) * PIX_ + p0, 0, 1);
  v8f acc = {};
  for (int kc = 0; kc < 2; ++kc) {
    __syncthreads();
    stage_f32<16, 32>(sA, w + o0 * 64 + kc * 32, 64);
#pragma unroll
    for (int i = threadIdx.x; i < 32 * 16; i += 32) {
      int k = i >> 4, n = i & 15;
      sB[(k >> 1) * 32 + 2 * n + (k & 1)] = hb[(size_t)(kc * 32 + k) * PIX_ + p0 + n];
    }
    __syncthreads();
    acc = wmma_bf16(frag_A(sA, 32), frag_B_pi(sB, 32), acc);
  }
  int l = threadIdx.x & 31, n = l & 15, mh = (l & 16) ? 8 : 0;
#pragma unroll
  for (int r = 0; r < 8; ++r) {
    int o = o0 + r + mh;
    hout[(size_t)(b * 64 + o) * PIX_ + p0 + n] = (__bf16)(acc[r] + bias[o]);
  }
}

// ---- inverse rDFT over W fused with skip-add + ReLU (in place, bf16) ------
__global__ void k_invW(const __bf16* __restrict__ Y, const __bf16* __restrict__ Wi,
                       __bf16* __restrict__ hout) {
  int blk = blockIdx.x;                             // 131072 blocks
  int w0 = (blk & 15) << 4;
  int h0 = ((blk >> 4) & 15) << 4;
  int bc = blk >> 8;
  v16bf a  = frag_A(Y + ((size_t)bc * 256 + h0) * 32, 32);
  v16bf bb = frag_B_pi(Wi + 2 * w0, 512);
  v8f acc = {};
  acc = wmma_bf16(a, bb, acc);
  __bf16* op = hout + (size_t)bc * PIX_;
  int l = threadIdx.x & 31, n = l & 15, mh = (l & 16) ? 8 : 0;
#pragma unroll
  for (int r = 0; r < 8; ++r) {
    size_t idx = (size_t)(h0 + r + mh) * 256 + w0 + n;
    float v = acc[r] + (float)op[idx];
    op[idx] = (__bf16)fmaxf(v, 0.f);
  }
}

// ---- head: fc1(128x64)+ReLU then fc2(3x128), 16 pixels / wave -------------
// Input tile staged LDS via GLOBAL_LOAD_ASYNC_TO_LDS_B32 (ASYNCcnt).
__global__ void k_head(const __bf16* __restrict__ h, const float* __restrict__ fc1_w,
                       const float* __restrict__ fc1_b, const float* __restrict__ fc2_w,
                       const float* __restrict__ fc2_b, float* __restrict__ out) {
  __shared__ __align__(16) __bf16 sB[64 * 16];      // input tile, row-major
  __shared__ __align__(16) __bf16 sA[16 * 128];     // weight staging
  __shared__ __align__(16) __bf16 sT[128 * 16];     // hidden acts, pair-interleaved
  int blk = blockIdx.x;                             // 32768 blocks
  int b = blk >> 12;
  int p0 = (blk & 4095) << 4;
  const __bf16* hb = h + (size_t)b * 64 * PIX_;
  // async raw copy: 64 rows x 32B -> LDS (b32 per lane, 16 issues)
  unsigned lds_base = (unsigned)(uintptr_t)(&sB[0]);
#pragma unroll
  for (int it = 0; it < 16; ++it) {
    int j = it * 32 + (int)(threadIdx.x & 31);      // dword index 0..511
    unsigned ldsa = lds_base + (unsigned)(j * 4);
    const void* g = (const char*)hb + ((size_t)(j >> 3) * PIX_ + p0) * 2 + (size_t)(j & 7) * 4;
    asm volatile("global_load_async_to_lds_b32 %0, %1, off" :: "v"(ldsa), "v"(g) : "memory");
  }
  asm volatile("s_wait_asynccnt 0" ::: "memory");
  __syncthreads();
  v16bf b0 = frag_B_rm(sB, 16);
  v16bf b1 = frag_B_rm(sB + 32 * 16, 16);
  int l = threadIdx.x & 31, n = l & 15, mh = (l & 16) ? 8 : 0;
  for (int mt = 0; mt < 8; ++mt) {
    __syncthreads();
    stage_f32<16, 64>(sA, fc1_w + mt * 16 * 64, 64);
    __syncthreads();
    v8f acc = {};
    acc = wmma_bf16(frag_A(sA, 64), b0, acc);
    acc = wmma_bf16(frag_A(sA + 32, 64), b1, acc);
#pragma unroll
    for (int r = 0; r < 8; ++r) {
      int m = mt * 16 + r + mh;
      float v = acc[r] + fc1_b[m];
      sT[(m >> 1) * 32 + 2 * n + (m & 1)] = (__bf16)fmaxf(v, 0.f);  // PI store
    }
  }
  __syncthreads();
#pragma unroll
  for (int i = threadIdx.x; i < 16 * 128; i += 32) {
    int m = i >> 7, k = i & 127;
    sA[i] = (m < 3) ? (__bf16)fc2_w[m * 128 + k] : (__bf16)0.f;
  }
  __syncthreads();
  v8f acc2 = {};
  for (int kc = 0; kc < 4; ++kc)
    acc2 = wmma_bf16(frag_A(sA + kc * 32, 128), frag_B_pi(sT + kc * 512, 32), acc2);
#pragma unroll
  for (int r = 0; r < 8; ++r) {
    int m = r + mh;
    if (m < 3)
      out[((size_t)b * 3 + m) * PIX_ + p0 + n] = acc2[r] + fc2_b[m];
  }
}

// ---------------------------------------------------------------------------
extern "C" void kernel_launch(void* const* d_in, const int* in_sizes, int n_in,
                              void* d_out, int out_size, void* d_ws, size_t ws_size,
                              hipStream_t stream) {
  (void)in_sizes; (void)n_in; (void)out_size; (void)ws_size;
  const float* x     = (const float*)d_in[0];
  const float* fc0_w = (const float*)d_in[1];
  const float* fc0_b = (const float*)d_in[2];
  const float* sw_re = (const float*)d_in[3];   // [4][64][64][16][16]
  const float* sw_im = (const float*)d_in[4];
  const float* w_w   = (const float*)d_in[5];   // [4][64][64]
  const float* w_b   = (const float*)d_in[6];   // [4][64]
  const float* fc1_w = (const float*)d_in[7];
  const float* fc1_b = (const float*)d_in[8];
  const float* fc2_w = (const float*)d_in[9];
  const float* fc2_b = (const float*)d_in[10];
  float* out = (float*)d_out;

  char* ws = (char*)d_ws;
  size_t off = 0;
  auto alloc = [&](size_t bytes) -> void* {
    void* p = ws + off;
    off = (off + bytes + 255) & ~(size_t)255;
    return p;
  };
  __bf16* hA  = (__bf16*)alloc((size_t)B_ * C_ * PIX_ * 2);  // 67 MB
  __bf16* hB  = (__bf16*)alloc((size_t)B_ * C_ * PIX_ * 2);  // 67 MB
  __bf16* G   = (__bf16*)alloc((size_t)B_ * C_ * 256 * 32 * 2);
  __bf16* Y   = (__bf16*)alloc((size_t)B_ * C_ * 256 * 32 * 2);
  float*  X   = (float*)alloc((size_t)B_ * C_ * 512 * 4);
  __bf16* Xh  = (__bf16*)alloc((size_t)B_ * C_ * 512 * 2);
  __bf16* Fwf = (__bf16*)alloc(8192 * 2);
  __bf16* Ch  = (__bf16*)alloc(4096 * 2);
  __bf16* Sh  = (__bf16*)alloc(4096 * 2);
  __bf16* Shn = (__bf16*)alloc(4096 * 2);
  __bf16* Hi1 = (__bf16*)alloc(8192 * 2);
  __bf16* Hi2 = (__bf16*)alloc(8192 * 2);
  __bf16* Wi  = (__bf16*)alloc(8192 * 2);

  k_init<<<1, 256, 0, stream>>>(Fwf, Ch, Sh, Shn, Hi1, Hi2, Wi);
  k_fc0<<<(B_ * C_ * PIX_) / 256, 256, 0, stream>>>(x, fc0_w, fc0_b, hA);

  __bf16* hin = hA;
  __bf16* hout = hB;
  for (int layer = 0; layer < 4; ++layer) {
    k_fwdW<<<(B_ * C_ * 256) / 16, 32, 0, stream>>>(hin, Fwf, G);
    k_fwdH<<<B_ * C_, 32, 0, stream>>>(G, Ch, Sh, Shn, X);
    k_mix<<<(B_ * C_ * 256) / 256, 256, 0, stream>>>(
        X, sw_re + (size_t)layer * 64 * 64 * 256, sw_im + (size_t)layer * 64 * 64 * 256, Xh);
    k_invH<<<B_ * C_ * 16, 32, 0, stream>>>(Xh, Hi1, Hi2, Y);
    k_conv<<<B_ * 4 * (PIX_ / 16), 32, 0, stream>>>(
        hin, w_w + (size_t)layer * 64 * 64, w_b + (size_t)layer * 64, hout);
    k_invW<<<B_ * C_ * 16 * 16, 32, 0, stream>>>(Y, Wi, hout);
    __bf16* t = hin; hin = hout; hout = t;
  }
  k_head<<<B_ * (PIX_ / 16), 32, 0, stream>>>(hin, fc1_w, fc1_b, fc2_w, fc2_b, out);
}
